// GPT2_20727512171023
// MI455X (gfx1250) — matrix operand order repaired and checked by reading, not compiled, not run on hardware
//
#include <hip/hip_runtime.h>
#include <math.h>

// ---- model dims ----
#define LAYERS 12
#define DMODEL 768
#define FFN    3072
#define VOCAB  50257
#define NHEAD  12
#define BATCH  2
#define SEQ    1024
#define DHEAD  64
#define EPSLN  1e-5f

typedef unsigned short bf16_t;
typedef __attribute__((ext_vector_type(16))) __bf16        v16bf;
typedef __attribute__((ext_vector_type(8)))  float         v8f;
typedef __attribute__((ext_vector_type(4)))  unsigned int  v4u;

union FragAB { v16bf v; v4u q[2]; };

__device__ __forceinline__ bf16_t f2bf(float f) {
  unsigned int u = __float_as_uint(f);
  u += 0x7fffu + ((u >> 16) & 1u);        // round-to-nearest-even
  return (bf16_t)(u >> 16);
}

// ---------------------------------------------------------------------------
// Embedding: x[b,s,:] = emb[ids[b,s],:] + pos_code[0, SEQ, :]  (ref indexes pos at Sln)
// ---------------------------------------------------------------------------
__global__ void __launch_bounds__(256)
embed_kernel(const int* __restrict__ ids, const float* __restrict__ emb,
             const float* __restrict__ pos, float* __restrict__ x)
{
  const int row = blockIdx.x;
  const int id  = ids[row];
  const float* e = emb + (size_t)id * DMODEL;
  const float* p = pos + (size_t)SEQ * DMODEL;
  float* xr = x + (size_t)row * DMODEL;
  for (int j = threadIdx.x; j < DMODEL; j += 256) xr[j] = e[j] + p[j];
}

// ---------------------------------------------------------------------------
// LayerNorm over D, writes bf16 (feeds WMMA GEMMs)
// ---------------------------------------------------------------------------
__global__ void __launch_bounds__(256)
layernorm_kernel(const float* __restrict__ x, const float* __restrict__ g,
                 const float* __restrict__ b, bf16_t* __restrict__ out)
{
  const int row = blockIdx.x, tid = threadIdx.x;
  __shared__ float red[256];
  const float* xr = x + (size_t)row * DMODEL;

  float s = 0.f;
  for (int j = tid; j < DMODEL; j += 256) s += xr[j];
  red[tid] = s; __syncthreads();
  for (int st = 128; st > 0; st >>= 1) { if (tid < st) red[tid] += red[tid + st]; __syncthreads(); }
  const float mean = red[0] / (float)DMODEL; __syncthreads();

  float v = 0.f;
  for (int j = tid; j < DMODEL; j += 256) { float d = xr[j] - mean; v += d * d; }
  red[tid] = v; __syncthreads();
  for (int st = 128; st > 0; st >>= 1) { if (tid < st) red[tid] += red[tid + st]; __syncthreads(); }
  const float rstd = rsqrtf(red[0] / (float)DMODEL + EPSLN);

  bf16_t* orow = out + (size_t)row * DMODEL;
  for (int j = tid; j < DMODEL; j += 256)
    orow[j] = f2bf((xr[j] - mean) * rstd * g[j] + b[j]);
}

// ---------------------------------------------------------------------------
// Convert f32 KxN -> bf16 transposed NxK (so WMMA B fragments load contiguously)
// ---------------------------------------------------------------------------
__global__ void __launch_bounds__(256)
cvt_transpose_kernel(const float* __restrict__ src, bf16_t* __restrict__ dst,
                     int K, int N)
{
  long long idx = (long long)blockIdx.x * 256 + threadIdx.x;
  if (idx >= (long long)K * N) return;
  int n = (int)(idx / K);
  int k = (int)(idx % K);
  dst[idx] = f2bf(src[(size_t)k * N + n]);
}

// ---------------------------------------------------------------------------
// v[b,s,h,dh] -> vT[b,h,dh,s]   (makes PV GEMM's B fragments contiguous)
// ---------------------------------------------------------------------------
__global__ void __launch_bounds__(256)
vtrans_kernel(const bf16_t* __restrict__ v, bf16_t* __restrict__ vT)
{
  int idx = blockIdx.x * 256 + threadIdx.x;
  if (idx >= BATCH * SEQ * DMODEL) return;
  int dh = idx % DHEAD;
  int hh = (idx / DHEAD) % NHEAD;
  int s  = (idx / DMODEL) % SEQ;
  int b  = idx / (DMODEL * SEQ);
  vT[((size_t)(b * NHEAD + hh) * DHEAD + dh) * SEQ + s] = v[idx];
}

// ---------------------------------------------------------------------------
// WMMA GEMM: C[M,N] = A[M,K](bf16,row) * BT[N,K](bf16,row)^T  (+bias,+res,+gelu)
// One wave computes a 32x32 macro-tile: 2 A frags + 2 B frags -> 4 WMMAs per
// 32-wide k-step (v_wmma_f32_16x16x32_bf16, fp32 accum).
// Edge handling (vocab N=50257): B column index is CLAMPED (no divergence in
// the loop); stores are masked on n < N. M is always a multiple of 32.
// Fragment layouts per CDNA5 ISA 7.12.2:
//   A lane: K = 16*(e/8) + 8*(lane>=16) + e%8 -> two contiguous 16B loads
//   B lane: K = 16*(e/8) + 8*(lane>=16) + e%8 -> same as A, two 16B loads
//   C/D:    VGPR j -> M = 8*(lane>=16) + j, N = lane%16
// ---------------------------------------------------------------------------
__global__ void __launch_bounds__(32)
gemm_wmma_kernel(const bf16_t* __restrict__ A,  int lda,
                 const bf16_t* __restrict__ BT, int ldbt,
                 const float*  __restrict__ bias,
                 const float*  __restrict__ residual, int ldres,
                 float*        __restrict__ Cf, int ldcf,
                 bf16_t*       __restrict__ Cb, int ldcb,
                 int M, int N, int K, int act)
{
  const int lane = threadIdx.x;
  const int lo   = lane & 15;
  const int hi   = lane >> 4;
  const int n0   = blockIdx.x * 32;
  const int m0   = blockIdx.y * 32;

  const int colB0 = min(n0 + lo,      N - 1);   // clamp: edge lanes recompute col N-1
  const int colB1 = min(n0 + 16 + lo, N - 1);

  const bf16_t* arow0 = A  + (size_t)(m0 + lo)      * lda + (hi ? 8 : 0);
  const bf16_t* arow1 = A  + (size_t)(m0 + 16 + lo) * lda + (hi ? 8 : 0);
  const bf16_t* brow0 = BT + (size_t)colB0 * ldbt + (hi ? 8 : 0);
  const bf16_t* brow1 = BT + (size_t)colB1 * ldbt + (hi ? 8 : 0);

  v8f acc00 = {0.f,0.f,0.f,0.f,0.f,0.f,0.f,0.f};
  v8f acc01 = acc00, acc10 = acc00, acc11 = acc00;

  for (int k0 = 0; k0 < K; k0 += 32) {
    FragAB a0, a1, b0, b1;
    a0.q[0] = *(const v4u*)(arow0 + k0);
    a0.q[1] = *(const v4u*)(arow0 + k0 + 16);
    a1.q[0] = *(const v4u*)(arow1 + k0);
    a1.q[1] = *(const v4u*)(arow1 + k0 + 16);
    b0.q[0] = *(const v4u*)(brow0 + k0);
    b0.q[1] = *(const v4u*)(brow0 + k0 + 16);
    b1.q[0] = *(const v4u*)(brow1 + k0);
    b1.q[1] = *(const v4u*)(brow1 + k0 + 16);
    acc00 = __builtin_amdgcn_wmma_f32_16x16x32_bf16(false, a0.v, false, b0.v, (short)0, acc00, false, false);
    acc01 = __builtin_amdgcn_wmma_f32_16x16x32_bf16(false, a0.v, false, b1.v, (short)0, acc01, false, false);
    acc10 = __builtin_amdgcn_wmma_f32_16x16x32_bf16(false, a1.v, false, b0.v, (short)0, acc10, false, false);
    acc11 = __builtin_amdgcn_wmma_f32_16x16x32_bf16(false, a1.v, false, b1.v, (short)0, acc11, false, false);
  }

  // epilogue: 2x2 sub-tiles
  v8f accs[4] = {acc00, acc01, acc10, acc11};
#pragma unroll
  for (int ti = 0; ti < 2; ++ti) {
#pragma unroll
    for (int tj = 0; tj < 2; ++tj) {
      const int n = n0 + tj * 16 + lo;
      if (n >= N) continue;
      const v8f av = accs[ti * 2 + tj];
      const int mb = m0 + ti * 16 + (hi ? 8 : 0);
      const float bn = bias ? bias[n] : 0.f;
#pragma unroll
      for (int j = 0; j < 8; ++j) {
        const int m = mb + j;
        float c = av[j] + bn;
        if (residual) c += residual[(size_t)m * ldres + n];
        if (act == 1) c = 0.5f * c * (1.0f + erff(c * 0.70710678118654752f)); // exact GELU
        if (Cf) Cf[(size_t)m * ldcf + n] = c;
        if (Cb) Cb[(size_t)m * ldcb + n] = f2bf(c);
      }
    }
  }
}

// ---------------------------------------------------------------------------
// Causal softmax over one (b,h): probs = softmax(scale*scores masked to -1e6)
// ---------------------------------------------------------------------------
__global__ void __launch_bounds__(256)
softmax_causal_kernel(const float* __restrict__ scores, bf16_t* __restrict__ probs,
                      float scale)
{
  const int row = blockIdx.x, tid = threadIdx.x;
  __shared__ float red[256];
  const float* srow = scores + (size_t)row * SEQ;

  float mx = -3.4e38f;
  for (int j = tid; j < SEQ; j += 256) {
    float v = (j <= row) ? srow[j] * scale : -1.0e6f;
    mx = fmaxf(mx, v);
  }
  red[tid] = mx; __syncthreads();
  for (int st = 128; st > 0; st >>= 1) { if (tid < st) red[tid] = fmaxf(red[tid], red[tid + st]); __syncthreads(); }
  mx = red[0]; __syncthreads();

  float sum = 0.f;
  for (int j = tid; j < SEQ; j += 256) {
    float v = (j <= row) ? srow[j] * scale : -1.0e6f;
    sum += expf(v - mx);
  }
  red[tid] = sum; __syncthreads();
  for (int st = 128; st > 0; st >>= 1) { if (tid < st) red[tid] += red[tid + st]; __syncthreads(); }
  const float inv = 1.0f / red[0];

  bf16_t* prow = probs + (size_t)row * SEQ;
  for (int j = tid; j < SEQ; j += 256) {
    float v = (j <= row) ? srow[j] * scale : -1.0e6f;
    prow[j] = f2bf(expf(v - mx) * inv);
  }
}

// ---------------------------------------------------------------------------
// Per-row NLL: lossrow[b*S+s] = -(logit[label] - max - log(sum exp)), s < S-1
// ---------------------------------------------------------------------------
__global__ void __launch_bounds__(256)
loss_row_kernel(const float* __restrict__ logits, const int* __restrict__ ids,
                float* __restrict__ lossrow)
{
  const int row = blockIdx.x, tid = threadIdx.x;
  const int b = row / SEQ, s = row % SEQ;
  if (s == SEQ - 1) { if (tid == 0) lossrow[row] = 0.f; return; }
  __shared__ float red[256];
  const float* lr = logits + (size_t)row * VOCAB;

  float mx = -3.4e38f;
  for (int j = tid; j < VOCAB; j += 256) mx = fmaxf(mx, lr[j]);
  red[tid] = mx; __syncthreads();
  for (int st = 128; st > 0; st >>= 1) { if (tid < st) red[tid] = fmaxf(red[tid], red[tid + st]); __syncthreads(); }
  mx = red[0]; __syncthreads();

  float sum = 0.f;
  for (int j = tid; j < VOCAB; j += 256) sum += expf(lr[j] - mx);
  red[tid] = sum; __syncthreads();
  for (int st = 128; st > 0; st >>= 1) { if (tid < st) red[tid] += red[tid + st]; __syncthreads(); }

  if (tid == 0) {
    const int label = ids[b * SEQ + s + 1];
    lossrow[row] = -(lr[label] - mx - logf(red[0]));
  }
}

__global__ void __launch_bounds__(256)
reduce_sum_kernel(const float* __restrict__ in, int n, float* __restrict__ out)
{
  __shared__ float red[256];
  const int tid = threadIdx.x;
  float s = 0.f;
  for (int j = tid; j < n; j += 256) s += in[j];
  red[tid] = s; __syncthreads();
  for (int st = 128; st > 0; st >>= 1) { if (tid < st) red[tid] += red[tid + st]; __syncthreads(); }
  if (tid == 0) out[0] = red[0];
}

// ---------------------------------------------------------------------------
extern "C" void kernel_launch(void* const* d_in, const int* in_sizes, int n_in,
                              void* d_out, int out_size, void* d_ws, size_t ws_size,
                              hipStream_t stream)
{
  (void)in_sizes; (void)n_in; (void)out_size; (void)ws_size;

  const int*   ids   = (const int*)  d_in[0];
  const float* emb   = (const float*)d_in[1];
  const float* pos   = (const float*)d_in[2];
  const float* ln1s  = (const float*)d_in[3];
  const float* ln1b  = (const float*)d_in[4];
  const float* wq    = (const float*)d_in[5];
  const float* bq    = (const float*)d_in[6];
  const float* wk    = (const float*)d_in[7];
  const float* bk    = (const float*)d_in[8];
  const float* wv    = (const float*)d_in[9];
  const float* bv    = (const float*)d_in[10];
  const float* wo    = (const float*)d_in[11];
  const float* bo    = (const float*)d_in[12];
  const float* ln2s  = (const float*)d_in[13];
  const float* ln2b  = (const float*)d_in[14];
  const float* w1    = (const float*)d_in[15];
  const float* b1    = (const float*)d_in[16];
  const float* w2    = (const float*)d_in[17];
  const float* b2    = (const float*)d_in[18];
  const float* lnfs  = (const float*)d_in[19];
  const float* lnfb  = (const float*)d_in[20];
  const float* wu    = (const float*)d_in[21];

  float* logits = (float*)d_out;
  float* loss   = logits + (size_t)BATCH * SEQ * VOCAB;

  // bump allocator over d_ws (256B aligned)
  char* wsp = (char*)d_ws;
  auto alloc = [&](size_t bytes) -> void* {
    void* p = (void*)wsp;
    wsp += (bytes + 255) & ~(size_t)255;
    return p;
  };
  const size_t ND = (size_t)BATCH * SEQ * DMODEL;
  float*  x    = (float*) alloc(ND * 4);
  bf16_t* hbuf = (bf16_t*)alloc(ND * 2);
  bf16_t* qb   = (bf16_t*)alloc(ND * 2);
  bf16_t* kb   = (bf16_t*)alloc(ND * 2);
  bf16_t* vb   = (bf16_t*)alloc(ND * 2);
  bf16_t* vT   = (bf16_t*)alloc(ND * 2);
  bf16_t* ab   = (bf16_t*)alloc(ND * 2);
  bf16_t* mbuf = (bf16_t*)alloc((size_t)BATCH * SEQ * FFN * 2);
  float*  sc   = (float*) alloc((size_t)SEQ * SEQ * 4);
  bf16_t* pb   = (bf16_t*)alloc((size_t)SEQ * SEQ * 2);
  bf16_t* wqT  = (bf16_t*)alloc((size_t)DMODEL * DMODEL * 2);
  bf16_t* wkT  = (bf16_t*)alloc((size_t)DMODEL * DMODEL * 2);
  bf16_t* wvT  = (bf16_t*)alloc((size_t)DMODEL * DMODEL * 2);
  bf16_t* woT  = (bf16_t*)alloc((size_t)DMODEL * DMODEL * 2);
  bf16_t* w1T  = (bf16_t*)alloc((size_t)DMODEL * FFN * 2);
  bf16_t* w2T  = (bf16_t*)alloc((size_t)DMODEL * FFN * 2);
  bf16_t* wuT  = (bf16_t*)alloc((size_t)VOCAB * DMODEL * 2);
  float*  lrow = (float*) alloc((size_t)BATCH * SEQ * 4);

  const int rows = BATCH * SEQ;                       // 2048
  embed_kernel<<<rows, 256, 0, stream>>>(ids, emb, pos, x);

  const int gDD = (DMODEL * DMODEL + 255) / 256;
  const int gDF = (DMODEL * FFN + 255) / 256;
  const dim3 gDmodel(DMODEL / 32, rows / 32, 1);      // (24,64)
  const dim3 gFFN(FFN / 32, rows / 32, 1);            // (96,64)
  const dim3 gScore(SEQ / 32, SEQ / 32, 1);           // (32,32)
  const dim3 gAV(DHEAD / 32, SEQ / 32, 1);            // (2,32)

  for (int l = 0; l < LAYERS; ++l) {
    const float* Wq = wq + (size_t)l * DMODEL * DMODEL;
    const float* Wk = wk + (size_t)l * DMODEL * DMODEL;
    const float* Wv = wv + (size_t)l * DMODEL * DMODEL;
    const float* Wo = wo + (size_t)l * DMODEL * DMODEL;
    const float* W1 = w1 + (size_t)l * DMODEL * FFN;
    const float* W2 = w2 + (size_t)l * DMODEL * FFN;

    cvt_transpose_kernel<<<gDD, 256, 0, stream>>>(Wq, wqT, DMODEL, DMODEL);
    cvt_transpose_kernel<<<gDD, 256, 0, stream>>>(Wk, wkT, DMODEL, DMODEL);
    cvt_transpose_kernel<<<gDD, 256, 0, stream>>>(Wv, wvT, DMODEL, DMODEL);
    cvt_transpose_kernel<<<gDD, 256, 0, stream>>>(Wo, woT, DMODEL, DMODEL);
    cvt_transpose_kernel<<<gDF, 256, 0, stream>>>(W1, w1T, DMODEL, FFN);
    cvt_transpose_kernel<<<gDF, 256, 0, stream>>>(W2, w2T, FFN, DMODEL);

    layernorm_kernel<<<rows, 256, 0, stream>>>(x, ln1s + l * DMODEL, ln1b + l * DMODEL, hbuf);

    gemm_wmma_kernel<<<gDmodel, 32, 0, stream>>>(hbuf, DMODEL, wqT, DMODEL, bq + l * DMODEL,
        nullptr, 0, nullptr, 0, qb, DMODEL, rows, DMODEL, DMODEL, 0);
    gemm_wmma_kernel<<<gDmodel, 32, 0, stream>>>(hbuf, DMODEL, wkT, DMODEL, bk + l * DMODEL,
        nullptr, 0, nullptr, 0, kb, DMODEL, rows, DMODEL, DMODEL, 0);
    gemm_wmma_kernel<<<gDmodel, 32, 0, stream>>>(hbuf, DMODEL, wvT, DMODEL, bv + l * DMODEL,
        nullptr, 0, nullptr, 0, vb, DMODEL, rows, DMODEL, DMODEL, 0);

    vtrans_kernel<<<(BATCH * SEQ * DMODEL + 255) / 256, 256, 0, stream>>>(vb, vT);

    for (int b = 0; b < BATCH; ++b) {
      for (int hh = 0; hh < NHEAD; ++hh) {
        const bf16_t* qh = qb + (size_t)b * SEQ * DMODEL + hh * DHEAD;
        const bf16_t* kh = kb + (size_t)b * SEQ * DMODEL + hh * DHEAD;
        const bf16_t* vh = vT + (size_t)(b * NHEAD + hh) * DHEAD * SEQ;
        bf16_t*       ah = ab + (size_t)b * SEQ * DMODEL + hh * DHEAD;

        // scores = q @ k^T   (A=q rows, BT=k rows; both lda=D)
        gemm_wmma_kernel<<<gScore, 32, 0, stream>>>(qh, DMODEL, kh, DMODEL, nullptr,
            nullptr, 0, sc, SEQ, nullptr, 0, SEQ, SEQ, DHEAD, 0);
        softmax_causal_kernel<<<SEQ, 256, 0, stream>>>(sc, pb, 0.125f);
        // a = probs @ v      (BT = vT[h] is DHxS row-major)
        gemm_wmma_kernel<<<gAV, 32, 0, stream>>>(pb, SEQ, vh, SEQ, nullptr,
            nullptr, 0, nullptr, 0, ah, DMODEL, SEQ, DHEAD, SEQ, 0);
      }
    }

    // x = x + a @ Wo + bo   (in-place residual: each element RMW by one lane)
    gemm_wmma_kernel<<<gDmodel, 32, 0, stream>>>(ab, DMODEL, woT, DMODEL, bo + l * DMODEL,
        x, DMODEL, x, DMODEL, nullptr, 0, rows, DMODEL, DMODEL, 0);

    layernorm_kernel<<<rows, 256, 0, stream>>>(x, ln2s + l * DMODEL, ln2b + l * DMODEL, hbuf);

    // m = gelu(h2 @ W1 + b1)
    gemm_wmma_kernel<<<gFFN, 32, 0, stream>>>(hbuf, DMODEL, w1T, DMODEL, b1 + l * FFN,
        nullptr, 0, nullptr, 0, mbuf, FFN, rows, FFN, DMODEL, 1);
    // x = x + m @ W2 + b2
    gemm_wmma_kernel<<<gDmodel, 32, 0, stream>>>(mbuf, FFN, w2T, FFN, b2 + l * DMODEL,
        x, DMODEL, x, DMODEL, nullptr, 0, rows, DMODEL, FFN, 0);
  }

  layernorm_kernel<<<rows, 256, 0, stream>>>(x, lnfs, lnfb, hbuf);

  {
    long long eu = (long long)DMODEL * VOCAB;
    cvt_transpose_kernel<<<(int)((eu + 255) / 256), 256, 0, stream>>>(wu, wuT, DMODEL, VOCAB);
  }
  const dim3 gU((VOCAB + 31) / 32, rows / 32, 1);     // (1571,64) — N edge clamped/masked
  gemm_wmma_kernel<<<gU, 32, 0, stream>>>(hbuf, DMODEL, wuT, DMODEL, nullptr,
      nullptr, 0, logits, VOCAB, nullptr, 0, rows, VOCAB, DMODEL, 0);

  loss_row_kernel<<<rows, 256, 0, stream>>>(logits, ids, lrow);
  reduce_sum_kernel<<<1, 256, 0, stream>>>(lrow, rows, loss);
}
